// CosineSimCodebook_66606352827341
// MI455X (gfx1250) — compile-verified, matrix operand order checked by reading, not run on hardware
//
#include <hip/hip_runtime.h>

#define DIM    256
#define NTOK   16384
#define KCB    8192
#define DECAY  0.8f
#define NSPLIT 4
#define NT_PER_SPLIT ((KCB / 16) / NSPLIT)   // 128 tiles of 16 codes

typedef __attribute__((ext_vector_type(8)))  __bf16 v8bf;
typedef __attribute__((ext_vector_type(16))) __bf16 v16bf;
typedef __attribute__((ext_vector_type(8)))  float  v8f;

__device__ __forceinline__ unsigned short f32_to_bf16_rne(float f) {
  unsigned int u = __float_as_uint(f);
  u += 0x7FFFu + ((u >> 16) & 1u);
  return (unsigned short)(u >> 16);
}

__device__ __forceinline__ v16bf cat8(v8bf lo, v8bf hi) {
  return __builtin_shufflevector(lo, hi, 0, 1, 2, 3, 4, 5, 6, 7,
                                         8, 9, 10, 11, 12, 13, 14, 15);
}

// ---------------------------------------------------------------------------
// Zero scratch (embed_sum + bins).
__global__ void vq_zero_kernel(float* __restrict__ p, int n) {
  int i = blockIdx.x * blockDim.x + threadIdx.x;
  if (i < n) p[i] = 0.0f;
}

// ---------------------------------------------------------------------------
// L2-normalize rows; emit f32 and bf16 copies. One wave32 per row, 8 rows/block.
__global__ void vq_normalize_kernel(const float* __restrict__ in,
                                    float* __restrict__ outf,
                                    unsigned short* __restrict__ outb,
                                    int rows) {
  const int wave = threadIdx.x >> 5;
  const int lane = threadIdx.x & 31;
  const int row  = blockIdx.x * 8 + wave;
  if (row >= rows) return;
  const float* p = in + (size_t)row * DIM;
  float v[8];
  float ss = 0.0f;
#pragma unroll
  for (int i = 0; i < 8; ++i) { v[i] = p[lane + i * 32]; ss += v[i] * v[i]; }
#pragma unroll
  for (int off = 16; off >= 1; off >>= 1) ss += __shfl_xor(ss, off, 32);
  const float inv = 1.0f / fmaxf(sqrtf(ss), 1e-12f);
  float*          q  = outf + (size_t)row * DIM;
  unsigned short* qb = outb + (size_t)row * DIM;
#pragma unroll
  for (int i = 0; i < 8; ++i) {
    const float t = v[i] * inv;
    q[lane + i * 32]  = t;
    qb[lane + i * 32] = f32_to_bf16_rne(t);
  }
}

// ---------------------------------------------------------------------------
// Fused (fn @ en^T) + partial row argmax via V_WMMA_F32_16X16X32_BF16.
// Block = 256 threads = 8 waves. Wave w owns M = 32 rows (two 16-row A tiles,
// register-resident -> each loaded B fragment feeds 2 WMMAs, halving operand
// feed bytes/FLOP). blockIdx.y selects one of NSPLIT codebook ranges; each
// block writes one (best,idx) candidate per row for a later reduce.
__global__ void __launch_bounds__(256) vq_dist_argmax_kernel(
    const unsigned short* __restrict__ fnb,   // [NTOK, DIM] bf16
    const unsigned short* __restrict__ enb,   // [KCB,  DIM] bf16
    float* __restrict__ pval,                 // [NSPLIT, NTOK]
    int* __restrict__ pidx)                   // [NSPLIT, NTOK]
{
  const int lane  = threadIdx.x & 31;
  const int wave  = threadIdx.x >> 5;
  const int m0    = blockIdx.x * 256 + wave * 32;
  const int half  = lane >> 4;                 // 0: K-low lanes, 1: K-high lanes
  const int kbase = half ? 8 : 0;
  const int ncol  = lane & 15;
  const int ns    = blockIdx.y;

  // Preload two A tiles (32x256 bf16) in ISA 16-bit A-matrix layout.
  v16bf A0[8], A1[8];
  {
    const unsigned short* a0 = fnb + (size_t)(m0 + (lane & 15)) * DIM;
    const unsigned short* a1 = a0 + 16 * DIM;
#pragma unroll
    for (int c = 0; c < 8; ++c) {
      const int kc = c * 32;
      A0[c] = cat8(*(const v8bf*)(a0 + kc + kbase),
                   *(const v8bf*)(a0 + kc + 16 + kbase));
      A1[c] = cat8(*(const v8bf*)(a1 + kc + kbase),
                   *(const v8bf*)(a1 + kc + 16 + kbase));
    }
  }

  float bv[2][8];
  int   bi[2][8];
#pragma unroll
  for (int tI = 0; tI < 2; ++tI)
#pragma unroll
    for (int i = 0; i < 8; ++i) { bv[tI][i] = -3.0e38f; bi[tI][i] = 0; }

  const int nt0 = ns * NT_PER_SPLIT;
  for (int t = 0; t < NT_PER_SPLIT; ++t) {
    const int nt = nt0 + t;
    const unsigned short* brow = enb + (size_t)(nt * 16 + ncol) * DIM;
    if (t + 1 < NT_PER_SPLIT)
      __builtin_prefetch(brow + 16 * DIM, 0, 0);

    v8f acc0 = {};
    v8f acc1 = {};
#pragma unroll
    for (int c = 0; c < 8; ++c) {
      const int kc = c * 32;
      v16bf B = cat8(*(const v8bf*)(brow + kc + kbase),
                     *(const v8bf*)(brow + kc + 16 + kbase));
      acc0 = __builtin_amdgcn_wmma_f32_16x16x32_bf16(
          false, A0[c], false, B, (short)0, acc0, false, false);
      acc1 = __builtin_amdgcn_wmma_f32_16x16x32_bf16(
          false, A1[c], false, B, (short)0, acc1, false, false);
    }

    const int ng = nt * 16 + ncol;
#pragma unroll
    for (int i = 0; i < 8; ++i) {
      const float v0 = acc0[i];
      if (v0 > bv[0][i]) { bv[0][i] = v0; bi[0][i] = ng; }
      const float v1 = acc1[i];
      if (v1 > bv[1][i]) { bv[1][i] = v1; bi[1][i] = ng; }
    }
  }

  // Reduce (max, first-index tie) across the 16 lanes holding one row's N slice.
#pragma unroll
  for (int tI = 0; tI < 2; ++tI) {
#pragma unroll
    for (int i = 0; i < 8; ++i) {
      float v   = bv[tI][i];
      int   idx = bi[tI][i];
#pragma unroll
      for (int off = 8; off >= 1; off >>= 1) {
        const float ov = __shfl_xor(v, off, 16);
        const int   oi = __shfl_xor(idx, off, 16);
        if (ov > v || (ov == v && oi < idx)) { v = ov; idx = oi; }
      }
      if ((lane & 15) == 0) {
        const int row = m0 + tI * 16 + i + (half ? 8 : 0);
        pval[(size_t)ns * NTOK + row] = v;
        pidx[(size_t)ns * NTOK + row] = idx;
      }
    }
  }
}

// ---------------------------------------------------------------------------
// Combine the NSPLIT partial argmax candidates per row (tie -> lowest index,
// i.e. first occurrence, matching jnp.argmax).
__global__ void vq_argmax_reduce_kernel(const float* __restrict__ pval,
                                        const int* __restrict__ pidx,
                                        int* __restrict__ ind,
                                        float* __restrict__ ind_f) {
  const int row = blockIdx.x * blockDim.x + threadIdx.x;
  if (row >= NTOK) return;
  float bvv = -3.0e38f;
  int   bii = 0x7fffffff;
#pragma unroll
  for (int s = 0; s < NSPLIT; ++s) {
    const float v = pval[(size_t)s * NTOK + row];
    const int   i = pidx[(size_t)s * NTOK + row];
    if (v > bvv || (v == bvv && i < bii)) { bvv = v; bii = i; }
  }
  ind[row]   = bii;
  ind_f[row] = (float)bii;
}

// ---------------------------------------------------------------------------
// quantize = embed[ind] gather; scatter fn into embed_sum/bins via f32 atomics.
// One block (256 threads) per token row.
__global__ void vq_gather_scatter_kernel(const float* __restrict__ embed,
                                         const float* __restrict__ fn,
                                         const int* __restrict__ ind,
                                         float* __restrict__ quant,
                                         float* __restrict__ esum,
                                         float* __restrict__ bins) {
  const int row = blockIdx.x;
  const int t   = threadIdx.x;
  const int k   = ind[row];
  quant[(size_t)row * DIM + t] = embed[(size_t)k * DIM + t];
  atomicAdd(&esum[(size_t)k * DIM + t], fn[(size_t)row * DIM + t]);
  if (t == 0) atomicAdd(&bins[k], 1.0f);
}

// ---------------------------------------------------------------------------
// embed_new = DECAY*embed + (1-DECAY)*(bins>0 ? l2norm(esum/bins) : en).
// One wave32 per code, 8 codes/block.
__global__ void vq_finalize_kernel(const float* __restrict__ embed,
                                   const float* __restrict__ enf,
                                   const float* __restrict__ esum,
                                   const float* __restrict__ bins,
                                   float* __restrict__ out_embed) {
  const int wave = threadIdx.x >> 5;
  const int lane = threadIdx.x & 31;
  const int k    = blockIdx.x * 8 + wave;
  const float b  = bins[k];
  const float binv = (b > 0.0f) ? (1.0f / b) : 1.0f;
  const float* s = esum + (size_t)k * DIM;
  float v[8];
  float ss = 0.0f;
#pragma unroll
  for (int i = 0; i < 8; ++i) { v[i] = s[lane + i * 32] * binv; ss += v[i] * v[i]; }
#pragma unroll
  for (int off = 16; off >= 1; off >>= 1) ss += __shfl_xor(ss, off, 32);
  const float inv = 1.0f / fmaxf(sqrtf(ss), 1e-12f);
  const float* e  = embed + (size_t)k * DIM;
  const float* en = enf + (size_t)k * DIM;
  float*       o  = out_embed + (size_t)k * DIM;
#pragma unroll
  for (int i = 0; i < 8; ++i) {
    const float nrm = (b > 0.0f) ? (v[i] * inv) : en[lane + i * 32];
    o[lane + i * 32] = DECAY * e[lane + i * 32] + (1.0f - DECAY) * nrm;
  }
}

// ---------------------------------------------------------------------------
extern "C" void kernel_launch(void* const* d_in, const int* in_sizes, int n_in,
                              void* d_out, int out_size, void* d_ws, size_t ws_size,
                              hipStream_t stream) {
  const float* x     = (const float*)d_in[0];   // [8,2048,256]
  const float* embed = (const float*)d_in[1];   // [8192,256]

  // Workspace layout (esum immediately followed by bins so one zero pass covers both).
  char* ws = (char*)d_ws;
  size_t off = 0;
  float* fn   = (float*)(ws + off); off += (size_t)NTOK * DIM * sizeof(float);
  float* en   = (float*)(ws + off); off += (size_t)KCB  * DIM * sizeof(float);
  float* esum = (float*)(ws + off); off += (size_t)KCB  * DIM * sizeof(float);
  float* bins = (float*)(ws + off); off += (size_t)KCB  * sizeof(float);
  int*   ind  = (int*)  (ws + off); off += (size_t)NTOK * sizeof(int);
  unsigned short* fnb = (unsigned short*)(ws + off); off += (size_t)NTOK * DIM * sizeof(unsigned short);
  unsigned short* enb = (unsigned short*)(ws + off); off += (size_t)KCB  * DIM * sizeof(unsigned short);
  float* pval = (float*)(ws + off); off += (size_t)NSPLIT * NTOK * sizeof(float);
  int*   pidx = (int*)  (ws + off); off += (size_t)NSPLIT * NTOK * sizeof(int);
  (void)ws_size; (void)in_sizes; (void)n_in; (void)out_size;

  // Output layout: quantize | embed_ind | embed_new (flat, f32).
  float* out_q   = (float*)d_out;
  float* out_ind = out_q + (size_t)NTOK * DIM;
  float* out_e   = out_ind + (size_t)NTOK;

  const int nzero = KCB * DIM + KCB;  // esum + bins (contiguous)
  vq_zero_kernel<<<(nzero + 255) / 256, 256, 0, stream>>>(esum, nzero);

  vq_normalize_kernel<<<NTOK / 8, 256, 0, stream>>>(x, fn, fnb, NTOK);
  vq_normalize_kernel<<<KCB / 8, 256, 0, stream>>>(embed, en, enb, KCB);

  dim3 dgrid(NTOK / 256, NSPLIT);
  vq_dist_argmax_kernel<<<dgrid, 256, 0, stream>>>(fnb, enb, pval, pidx);

  vq_argmax_reduce_kernel<<<(NTOK + 255) / 256, 256, 0, stream>>>(pval, pidx, ind, out_ind);

  vq_gather_scatter_kernel<<<NTOK, 256, 0, stream>>>(embed, fn, ind, out_q, esum, bins);

  vq_finalize_kernel<<<KCB / 8, 256, 0, stream>>>(embed, en, esum, bins, out_e);
}
